// ServerModelViT_30081950941641
// MI455X (gfx1250) — matrix operand order repaired and checked by reading, not compiled, not run on hardware
//
#include <hip/hip_runtime.h>
#include <hip/hip_bf16.h>

typedef unsigned short u16;
typedef unsigned int   u32;
typedef __bf16 bf16;
typedef bf16  v16bf  __attribute__((ext_vector_type(16)));
typedef float v8f    __attribute__((ext_vector_type(8)));
typedef u32   u32x8  __attribute__((ext_vector_type(8)));

#define EMB 768
#define HEADS 8
#define DH 96
#define DEPTH 12
#define FF 3072
#define NCLS 1000
#define BATCH 32
#define NTOK 197
#define NPAD 208                  // 197 padded to multiple of 16 (16B-aligned bf16 rows)
#define ROWS (BATCH * NTOK)       // 6304
#define ZB (BATCH * HEADS)        // 256

__device__ __forceinline__ u16 f2bf(float f) {
    u32 u = __float_as_uint(f);
    u32 r = (u + 0x7FFFu + ((u >> 16) & 1u)) >> 16;
    return (u16)r;
}

// CDNA5 async global->LDS copy (ASYNCcnt-tracked). lds_off = LDS byte offset.
__device__ __forceinline__ void async_ld_b128(u32 lds_off, const void* gptr) {
    asm volatile("global_load_async_to_lds_b128 %0, %1, off"
                 :: "v"(lds_off), "v"(gptr) : "memory");
}
__device__ __forceinline__ void wait_async0() {
    asm volatile("s_wait_asynccnt 0x0" ::: "memory");
}
__device__ __forceinline__ u32 lds_off_of(const void* p) {
    return (u32)(size_t)p;   // generic ptr low 32 bits == LDS offset
}
__device__ __forceinline__ void zero16B(u16* p) {
    u32* q = (u32*)p;
    q[0] = 0; q[1] = 0; q[2] = 0; q[3] = 0;
}

// ---------------------------------------------------------------------------
// Tiled GEMM, compile-time specialized:  out = act(A(bf16) @ B + bias) + resid
// BM=BN=128, BK=32, 256 threads (8 waves; wave = 64 rows x 32 cols = 4x2 WMMA).
// A bf16 row-major [M,K]; REQUIRES K%32==0 and lda%8==0 (pitches padded so).
// B: BRM -> bf16 [N,K] (ldb%8==0); !BRM -> f32 [K,N] weights (ldb%4==0),
// transposed+converted to bf16 during staging. Double-buffered LDS: async
// global->LDS copies for tile i+1 overlap the WMMAs on tile i.
// Output offset (z/cZmod)*cOuter + (z%cZmod)*cInner supports head scatter.
// ---------------------------------------------------------------------------
template<bool BRM, bool OUTBF, bool GELU, bool HASBIAS, bool HASRES>
__global__ __launch_bounds__(256) void gemm_wmma(
    const u16* __restrict__ A, int lda, long long sA,
    const void* __restrict__ B, int ldb, long long sB,
    const float* __restrict__ bias,
    const float* __restrict__ resid, int ldr,
    void* __restrict__ out, int ldc,
    long long cOuter, long long cInner, int cZmod,
    int M, int N, int K)
{
    __shared__ u16 As[2][128][40];   // [buf][m][k], 80B rows (16B aligned)
    __shared__ u16 Bs[2][128][40];   // [buf][n][k]

    const int z = blockIdx.z;
    A += (long long)z * sA;
    const float* Bf = (const float*)B + (long long)z * sB;
    const u16*   Bh = (const u16*)B   + (long long)z * sB;
    const long long co = (long long)(z / cZmod) * cOuter + (long long)(z % cZmod) * cInner;

    const int tile_n0 = blockIdx.x * 128;
    const int tile_m0 = blockIdx.y * 128;
    const int tid  = threadIdx.x;
    const int wave = tid >> 5;
    const int lane = tid & 31;
    const int wm = wave >> 2;        // 0..1  -> 64 rows each
    const int wn = wave & 3;         // 0..3  -> 32 cols each
    const int lhalf = lane >> 4;     // 0/1
    const int l16   = lane & 15;

    // ---- tile staging (chunk-granular edge guards only) ----
    auto stageA = [&](int buf, int k0) {
        #pragma unroll
        for (int c = 0; c < 2; c++) {
            int chunk = tid + (c << 8);
            int r  = chunk >> 2;           // 0..127
            int cg = (chunk & 3) << 3;     // 0,8,16,24
            int gm = tile_m0 + r;
            if (gm < M)
                async_ld_b128(lds_off_of(&As[buf][r][cg]),
                              A + (long long)gm * lda + k0 + cg);
            else
                zero16B(&As[buf][r][cg]);
        }
    };
    auto stageB = [&](int buf, int k0) {
        if constexpr (BRM) {
            #pragma unroll
            for (int c = 0; c < 2; c++) {
                int chunk = tid + (c << 8);
                int r  = chunk >> 2;
                int cg = (chunk & 3) << 3;
                int gn = tile_n0 + r;
                if (gn < N)
                    async_ld_b128(lds_off_of(&Bs[buf][r][cg]),
                                  Bh + (long long)gn * ldb + k0 + cg);
                else
                    zero16B(&Bs[buf][r][cg]);
            }
        } else {
            // f32 [K,N] weights -> bf16 Bs[n][k]; 16 k-pairs x 32 n-groups(4)
            #pragma unroll
            for (int c = 0; c < 2; c++) {
                int item = tid + (c << 8);
                int kp = item >> 5;            // 0..15 -> k = 2*kp
                int ng = (item & 31) << 2;     // 0..124
                int k  = 2 * kp;
                const float* g0 = Bf + (long long)(k0 + k) * ldb + tile_n0 + ng;
                if (tile_n0 + ng + 3 < N) {
                    float4 a = *(const float4*)g0;
                    float4 b = *(const float4*)(g0 + ldb);
                    *(u32*)&Bs[buf][ng + 0][k] = (u32)f2bf(a.x) | ((u32)f2bf(b.x) << 16);
                    *(u32*)&Bs[buf][ng + 1][k] = (u32)f2bf(a.y) | ((u32)f2bf(b.y) << 16);
                    *(u32*)&Bs[buf][ng + 2][k] = (u32)f2bf(a.z) | ((u32)f2bf(b.z) << 16);
                    *(u32*)&Bs[buf][ng + 3][k] = (u32)f2bf(a.w) | ((u32)f2bf(b.w) << 16);
                } else {
                    #pragma unroll
                    for (int e = 0; e < 4; e++) {
                        int gn = tile_n0 + ng + e;
                        u32 lo = 0, hi = 0;
                        if (gn < N) {
                            lo = f2bf(g0[e]);
                            hi = f2bf(g0[ldb + e]);
                        }
                        *(u32*)&Bs[buf][ng + e][k] = lo | (hi << 16);
                    }
                }
            }
        }
    };

    v8f acc[4][2] = {};

    stageA(0, 0);
    stageB(0, 0);
    const int nk = K >> 5;
    for (int it = 0; it < nk; it++) {
        wait_async0();          // own async copies into current buffer done
        __syncthreads();        // all waves' staging of current buffer done

        const int cur = it & 1;
        if (it + 1 < nk) {      // prefetch next tile into the other buffer
            stageA(1 - cur, (it + 1) << 5);
            stageB(1 - cur, (it + 1) << 5);
        }

        // --- B fragments (32x16): elem e -> k=(lane>>4)*16+e, n=lane&15
        v16bf bfr[2];
        #pragma unroll
        for (int j = 0; j < 2; j++) {
            const u16* bp = &Bs[cur][wn * 32 + j * 16 + l16][lhalf * 16];
            u32x8 t;
            #pragma unroll
            for (int v = 0; v < 8; v++) t[v] = *(const u32*)(bp + 2 * v);
            bfr[j] = __builtin_bit_cast(v16bf, t);
        }
        // --- A fragments (16x32): elem e -> k=(lane>>4)*16+e
        #pragma unroll
        for (int i = 0; i < 4; i++) {
            const u16* ap = &As[cur][wm * 64 + i * 16 + l16][0];
            const int kb = lhalf * 16;
            u32x8 t;
            #pragma unroll
            for (int v = 0; v < 8; v++) {
                int e = 2 * v;
                int k = kb + e;
                t[v] = *(const u32*)(ap + k);
            }
            v16bf af = __builtin_bit_cast(v16bf, t);
            #pragma unroll
            for (int j = 0; j < 2; j++)
                acc[i][j] = __builtin_amdgcn_wmma_f32_16x16x32_bf16(
                    false, af, false, bfr[j], (short)0, acc[i][j], false, false);
        }
    }

    // --- epilogue: C layout VGPR r -> m = base + r + (lane>=16 ? 8 : 0)
    #pragma unroll
    for (int i = 0; i < 4; i++) {
        #pragma unroll
        for (int j = 0; j < 2; j++) {
            #pragma unroll
            for (int r = 0; r < 8; r++) {
                int m = tile_m0 + wm * 64 + i * 16 + r + (lhalf << 3);
                int n = tile_n0 + wn * 32 + j * 16 + l16;
                if (m < M && n < N) {
                    float v = acc[i][j][r];
                    if constexpr (HASBIAS) v += bias[n];
                    if constexpr (GELU)
                        v = 0.5f * v * (1.0f + erff(v * 0.70710678118654752f));
                    if constexpr (HASRES) v += resid[(long long)m * ldr + n];
                    long long o = co + (long long)m * ldc + n;
                    if constexpr (OUTBF) ((u16*)out)[o] = f2bf(v);
                    else                 ((float*)out)[o] = v;
                }
            }
        }
    }
}

// ---------------------------------------------------------------------------
// LayerNorm over last dim -> bf16 out. One block per row.
// ---------------------------------------------------------------------------
__global__ __launch_bounds__(256) void ln_bf16_kernel(
    const float* __restrict__ x, const float* __restrict__ g,
    const float* __restrict__ bta, u16* __restrict__ out, int ncols)
{
    __shared__ float s1[256], s2[256];
    const long long row = blockIdx.x;
    const float* xr = x + row * ncols;
    float a = 0.f, b = 0.f;
    for (int c = threadIdx.x; c < ncols; c += 256) {
        float v = xr[c]; a += v; b += v * v;
    }
    s1[threadIdx.x] = a; s2[threadIdx.x] = b;
    __syncthreads();
    for (int off = 128; off > 0; off >>= 1) {
        if (threadIdx.x < off) {
            s1[threadIdx.x] += s1[threadIdx.x + off];
            s2[threadIdx.x] += s2[threadIdx.x + off];
        }
        __syncthreads();
    }
    float inv_n = 1.0f / (float)ncols;
    float mu = s1[0] * inv_n;
    float var = s2[0] * inv_n - mu * mu;
    float inv = rsqrtf(var + 1e-5f);
    for (int c = threadIdx.x; c < ncols; c += 256)
        out[row * ncols + c] = f2bf((xr[c] - mu) * inv * g[c] + bta[c]);
}

// ---------------------------------------------------------------------------
// Row softmax (len=197) then multiply by 1/sqrt(E) AFTER softmax (faithful).
// Output bf16 with row pitch NPAD, zero padded cols [len, NPAD).
// ---------------------------------------------------------------------------
__global__ __launch_bounds__(256) void softmax_bf16_kernel(
    const float* __restrict__ s, u16* __restrict__ att, int len, float scale)
{
    __shared__ float red[256];
    const long long row = blockIdx.x;
    const float* sr = s + row * len;
    float mx = -3.4e38f;
    for (int c = threadIdx.x; c < len; c += 256) mx = fmaxf(mx, sr[c]);
    red[threadIdx.x] = mx; __syncthreads();
    for (int off = 128; off > 0; off >>= 1) {
        if (threadIdx.x < off)
            red[threadIdx.x] = fmaxf(red[threadIdx.x], red[threadIdx.x + off]);
        __syncthreads();
    }
    mx = red[0]; __syncthreads();
    float sum = 0.f;
    for (int c = threadIdx.x; c < len; c += 256) sum += expf(sr[c] - mx);
    red[threadIdx.x] = sum; __syncthreads();
    for (int off = 128; off > 0; off >>= 1) {
        if (threadIdx.x < off) red[threadIdx.x] += red[threadIdx.x + off];
        __syncthreads();
    }
    float m = scale / red[0];
    for (int c = threadIdx.x; c < NPAD; c += 256) {
        u16 v = 0;
        if (c < len) v = f2bf(expf(sr[c] - mx) * m);
        att[row * NPAD + c] = v;
    }
}

// ---------------------------------------------------------------------------
// Split fused QKV (inner layout [head][dh][3]) into per-head bf16 Q, K and
// Vᵀ (pitch NPAD, zero-padded keys so the AV GEMM can run K=NPAD unguarded).
// ---------------------------------------------------------------------------
__global__ __launch_bounds__(256) void split_qkv_kernel(
    const float* __restrict__ qkv, u16* __restrict__ qh,
    u16* __restrict__ kh, u16* __restrict__ vt)
{
    long long idx = (long long)blockIdx.x * 256 + threadIdx.x;
    const long long total = (long long)ZB * NPAD * DH;
    if (idx >= total) return;
    int d  = idx % DH;
    int np = (idx / DH) % NPAD;
    int z  = idx / ((long long)DH * NPAD);
    if (np < NTOK) {
        int b = z / HEADS, h = z % HEADS;
        long long base = ((long long)b * NTOK + np) * (3 * EMB) + h * (DH * 3) + d * 3;
        qh[((long long)z * NTOK + np) * DH + d] = f2bf(qkv[base + 0]);
        kh[((long long)z * NTOK + np) * DH + d] = f2bf(qkv[base + 1]);
        vt[((long long)z * DH + d) * NPAD + np] = f2bf(qkv[base + 2]);
    } else {
        vt[((long long)z * DH + d) * NPAD + np] = 0;
    }
}

__global__ __launch_bounds__(256) void copy_f32_kernel(
    const float* __restrict__ src, float* __restrict__ dst, long long n)
{
    long long i = (long long)blockIdx.x * 256 + threadIdx.x;
    if (i < n) dst[i] = src[i];
}

__global__ __launch_bounds__(256) void meanpool_kernel(
    const float* __restrict__ x, float* __restrict__ pooled)
{
    int idx = blockIdx.x * 256 + threadIdx.x;
    if (idx >= BATCH * EMB) return;
    int b = idx / EMB, e = idx % EMB;
    const float* p = x + (long long)b * NTOK * EMB + e;
    float s = 0.f;
    for (int n = 0; n < NTOK; n++) s += p[(long long)n * EMB];
    pooled[idx] = s * (1.0f / (float)NTOK);
}

// ---------------------------------------------------------------------------
extern "C" void kernel_launch(void* const* d_in, const int* in_sizes, int n_in,
                              void* d_out, int out_size, void* d_ws, size_t ws_size,
                              hipStream_t stream)
{
    const float* x      = (const float*)d_in[0];
    const float* ln1_g  = (const float*)d_in[1];
    const float* ln1_b  = (const float*)d_in[2];
    const float* qkv_w  = (const float*)d_in[3];
    const float* qkv_b  = (const float*)d_in[4];
    const float* proj_w = (const float*)d_in[5];
    const float* proj_b = (const float*)d_in[6];
    const float* ln2_g  = (const float*)d_in[7];
    const float* ln2_b  = (const float*)d_in[8];
    const float* ff1_w  = (const float*)d_in[9];
    const float* ff1_b  = (const float*)d_in[10];
    const float* ff2_w  = (const float*)d_in[11];
    const float* ff2_b  = (const float*)d_in[12];
    const float* cls_ln_g = (const float*)d_in[13];
    const float* cls_ln_b = (const float*)d_in[14];
    const float* cls_w  = (const float*)d_in[15];
    const float* cls_b  = (const float*)d_in[16];

    // --- workspace layout (256B aligned) ---
    char* w = (char*)d_ws;
    size_t off = 0;
    auto take = [&](size_t bytes) {
        void* p = w + off;
        off += (bytes + 255) & ~(size_t)255;
        return p;
    };
    float* xcur  = (float*)take((size_t)ROWS * EMB * 4);
    u16*   hbuf  = (u16*)  take((size_t)ROWS * EMB * 2);
    float* qkvb  = (float*)take((size_t)ROWS * 3 * EMB * 4);
    u16*   qh    = (u16*)  take((size_t)ZB * NTOK * DH * 2);
    u16*   kh    = (u16*)  take((size_t)ZB * NTOK * DH * 2);
    u16*   vt    = (u16*)  take((size_t)ZB * DH * NPAD * 2);
    float* sbuf  = (float*)take((size_t)ZB * NTOK * NTOK * 4);
    u16*   attb  = (u16*)  take((size_t)ZB * NTOK * NPAD * 2);
    u16*   obuf  = (u16*)  take((size_t)ROWS * EMB * 2);
    u16*   ffbuf = (u16*)  take((size_t)ROWS * FF * 2);
    float* pooled  = (float*)take((size_t)BATCH * EMB * 4);
    u16*   pooledh = (u16*)  take((size_t)BATCH * EMB * 2);

    const float inv_scaling = 1.0f / sqrtf((float)EMB);  // applied AFTER softmax

    // residual chain lives in xcur (never mutate d_in)
    {
        long long n = (long long)ROWS * EMB;
        copy_f32_kernel<<<(unsigned)((n + 255) / 256), 256, 0, stream>>>(x, xcur, n);
    }

    for (int l = 0; l < DEPTH; l++) {
        const float* l1g = ln1_g + l * EMB;
        const float* l1b = ln1_b + l * EMB;
        const float* qw  = qkv_w + (long long)l * EMB * 3 * EMB;
        const float* qb  = qkv_b + l * 3 * EMB;
        const float* pw  = proj_w + (long long)l * EMB * EMB;
        const float* pb  = proj_b + l * EMB;
        const float* l2g = ln2_g + l * EMB;
        const float* l2b = ln2_b + l * EMB;
        const float* f1w = ff1_w + (long long)l * EMB * FF;
        const float* f1b = ff1_b + l * FF;
        const float* f2w = ff2_w + (long long)l * FF * EMB;
        const float* f2b = ff2_b + l * EMB;

        // LN1 -> hbuf (bf16)
        ln_bf16_kernel<<<ROWS, 256, 0, stream>>>(xcur, l1g, l1b, hbuf, EMB);

        // QKV GEMM: [6304,768] @ f32[768,2304] + b -> qkvb (f32)
        gemm_wmma<false, false, false, true, false><<<dim3(18, 50, 1), 256, 0, stream>>>(
            hbuf, EMB, 0, qw, 3 * EMB, 0, qb, nullptr, 0,
            qkvb, 3 * EMB, 0, 0, 1, ROWS, 3 * EMB, EMB);

        // split heads -> qh, kh, vt (bf16; vt zero-padded to NPAD keys)
        {
            long long n = (long long)ZB * NPAD * DH;
            split_qkv_kernel<<<(unsigned)((n + 255) / 256), 256, 0, stream>>>(qkvb, qh, kh, vt);
        }

        // S = Q @ Kᵀ per (b,h): M=N=197, K=96, batched over z=256
        gemm_wmma<true, false, false, false, false><<<dim3(2, 2, ZB), 256, 0, stream>>>(
            qh, DH, (long long)NTOK * DH,
            kh, DH, (long long)NTOK * DH,
            nullptr, nullptr, 0,
            sbuf, NTOK, (long long)NTOK * NTOK, 0, 1,
            NTOK, NTOK, DH);

        // softmax rows, post-scale by 1/sqrt(E) -> attb (bf16, pitch NPAD)
        softmax_bf16_kernel<<<ZB * NTOK, 256, 0, stream>>>(sbuf, attb, NTOK, inv_scaling);

        // O = Att @ V: M=197, N=96, K=NPAD (zero-padded); scatter heads into obuf[b,n,h*d]
        gemm_wmma<true, true, false, false, false><<<dim3(1, 2, ZB), 256, 0, stream>>>(
            attb, NPAD, (long long)NTOK * NPAD,
            vt, NPAD, (long long)DH * NPAD,
            nullptr, nullptr, 0,
            obuf, EMB, (long long)NTOK * EMB, DH, HEADS,
            NTOK, DH, NPAD);

        // proj GEMM + bias + residual -> xcur (in place; disjoint RMW per element)
        gemm_wmma<false, false, false, true, true><<<dim3(6, 50, 1), 256, 0, stream>>>(
            obuf, EMB, 0, pw, EMB, 0, pb, xcur, EMB,
            xcur, EMB, 0, 0, 1, ROWS, EMB, EMB);

        // LN2 -> hbuf
        ln_bf16_kernel<<<ROWS, 256, 0, stream>>>(xcur, l2g, l2b, hbuf, EMB);

        // FF1 GEMM + bias + exact GELU -> ffbuf (bf16)
        gemm_wmma<false, true, true, true, false><<<dim3(24, 50, 1), 256, 0, stream>>>(
            hbuf, EMB, 0, f1w, FF, 0, f1b, nullptr, 0,
            ffbuf, FF, 0, 0, 1, ROWS, FF, EMB);

        // FF2 GEMM + bias + residual -> xcur
        gemm_wmma<false, false, false, true, true><<<dim3(6, 50, 1), 256, 0, stream>>>(
            ffbuf, FF, 0, f2w, EMB, 0, f2b, xcur, EMB,
            xcur, EMB, 0, 0, 1, ROWS, EMB, FF);
    }

    // mean pool over tokens -> pooled; LN; classifier head -> d_out
    meanpool_kernel<<<(BATCH * EMB + 255) / 256, 256, 0, stream>>>(xcur, pooled);
    ln_bf16_kernel<<<BATCH, 256, 0, stream>>>(pooled, cls_ln_g, cls_ln_b, pooledh, EMB);
    gemm_wmma<false, false, false, true, false><<<dim3(8, 1, 1), 256, 0, stream>>>(
        pooledh, EMB, 0, cls_w, NCLS, 0, cls_b, nullptr, 0,
        d_out, NCLS, 0, 0, 1, BATCH, NCLS, EMB);
}
